// DynamicRoutingLayer_28209345200812
// MI455X (gfx1250) — matrix operand order repaired and checked by reading, not compiled, not run on hardware
//
#include <hip/hip_runtime.h>
#include <hip/hip_bf16.h>

typedef __bf16 bf16;
typedef bf16  bf16x8 __attribute__((ext_vector_type(8)));
typedef bf16  v16bf  __attribute__((ext_vector_type(16)));
typedef float v8f    __attribute__((ext_vector_type(8)));

#define NB    128
#define MSEQ  512
#define DIMSZ 128
#define TM    64
#define TILES (MSEQ / TM)   // 8

// LDS layout (bytes): S[64][512] f32 @0 (131072), aliased by WbT[128][128] bf16 (32768)
// Xbf[512][128] bf16 @131072 (131072), Ybf[64][128] bf16 @262144 (16384),
// rowmax[64] @278528, rowsum[64] @278784  -> total 279040
#define SMEM_BYTES 279040

// d_ws layout: G f32 [128][512] @0 (262144), WbT bf16 [128][128] @262144 (32768),
// x_bf16 [128*512*128] @294912 (16777216)  -> need 17072128 bytes
#define WS_G_OFF   0
#define WS_WBT_OFF 262144
#define WS_XB_OFF  294912
#define WS_NEED    17072128ull

#define K_W_SCALE (0.25f * 0.08838834764831845f)  // (1/H) * dim^-0.5 folded into Wbar

#if __has_builtin(__builtin_amdgcn_global_load_async_to_lds_b128)
#define HAVE_ASYNC_LDS 1
typedef int v4i_ __attribute__((ext_vector_type(4)));
typedef __attribute__((address_space(1))) v4i_* g_v4i_p;   // global src (AS1)
typedef __attribute__((address_space(3))) v4i_* l_v4i_p;   // LDS dst (AS3)
#else
#define HAVE_ASYNC_LDS 0
#endif

static __device__ __forceinline__ void wait_async_zero() {
#if HAVE_ASYNC_LDS
#if __has_builtin(__builtin_amdgcn_s_wait_asynccnt)
  __builtin_amdgcn_s_wait_asynccnt(0);
#else
  asm volatile("s_wait_asynccnt 0" ::: "memory");
#endif
#endif
}

static __device__ __forceinline__ v16bf ld_frag(const bf16* lo, const bf16* hi) {
  bf16x8 a = *reinterpret_cast<const bf16x8*>(lo);
  bf16x8 b = *reinterpret_cast<const bf16x8*>(hi);
  v16bf r;
#pragma unroll
  for (int i = 0; i < 8; ++i) { r[i] = a[i]; r[i + 8] = b[i]; }
  return r;
}

static __device__ __forceinline__ v8f wmma_bf16(v16bf a, v16bf b, v8f c) {
  // D = A(16x32 bf16) * B(32x16 bf16) + C(16x16 f32)
  return __builtin_amdgcn_wmma_f32_16x16x32_bf16(false, a, false, b, (short)0, c, false, false);
}

// ---- prep: x f32 -> bf16 (whole tensor, once per call)
__global__ __launch_bounds__(256)
void convert_x_kernel(const float* __restrict__ x, bf16* __restrict__ xb) {
  const size_t i = (size_t)blockIdx.x * 256 + threadIdx.x;   // over N*M*D/4
  const float4 v = reinterpret_cast<const float4*>(x)[i];
  union { bf16 h[4]; uint2 u; } pk;
  pk.h[0] = (bf16)v.x; pk.h[1] = (bf16)v.y; pk.h[2] = (bf16)v.z; pk.h[3] = (bf16)v.w;
  reinterpret_cast<uint2*>(xb)[i] = pk.u;
}

// ---- prep: WbT[f][e] = (scale/H) * sum_h rw[h][e][f]
__global__ __launch_bounds__(256)
void build_wbt_kernel(const float* __restrict__ rw, bf16* __restrict__ wbt) {
  const int i = blockIdx.x * 256 + threadIdx.x;   // 16384
  const int e = i >> 7, f = i & 127;
  float s = 0.f;
#pragma unroll
  for (int h = 0; h < 4; ++h) s += rw[((h * DIMSZ) + e) * DIMSZ + f];
  wbt[f * DIMSZ + e] = (bf16)(s * K_W_SCALE);
}

__global__ __launch_bounds__(256)
void routing_scores_kernel(const float* __restrict__ x,
                           const float* __restrict__ rw,
                           const bf16* __restrict__ xpre,
                           const bf16* __restrict__ wpre,
                           float* __restrict__ G,
                           int use_pre) {
  extern __shared__ char smem[];
  float* S      = reinterpret_cast<float*>(smem);            // [64][512]
  bf16*  WbT    = reinterpret_cast<bf16*>(smem);             // alias: [128][128] (f, e)
  bf16*  Xbf    = reinterpret_cast<bf16*>(smem + 131072);    // [512][128]
  bf16*  Ybf    = reinterpret_cast<bf16*>(smem + 262144);    // [64][128]
  float* rowmax = reinterpret_cast<float*>(smem + 278528);   // [64]
  float* rowsum = reinterpret_cast<float*>(smem + 278784);   // [64]

  const int n    = blockIdx.x / TILES;
  const int mt   = blockIdx.x % TILES;
  const int tid  = threadIdx.x;
  const int lane = tid & 31;
  const int wave = tid >> 5;       // 0..7

  // ---- Stage 0: fill Xbf and WbT
  if (use_pre) {
    const bf16* xnb = xpre + (size_t)n * MSEQ * DIMSZ;
#if HAVE_ASYNC_LDS
    // gfx1250 async data mover path: global -> LDS without VGPR staging (ASYNCcnt)
    for (int i = tid; i < (MSEQ * DIMSZ) / 8; i += 256)      // 16B chunks
      __builtin_amdgcn_global_load_async_to_lds_b128(
          (g_v4i_p)(void*)(xnb + i * 8), (l_v4i_p)(void*)(Xbf + i * 8), 0, 0);
    for (int i = tid; i < (DIMSZ * DIMSZ) / 8; i += 256)
      __builtin_amdgcn_global_load_async_to_lds_b128(
          (g_v4i_p)(void*)(wpre + i * 8), (l_v4i_p)(void*)(WbT + i * 8), 0, 0);
    wait_async_zero();
#else
    for (int i = tid; i < (MSEQ * DIMSZ) / 8; i += 256)
      reinterpret_cast<uint4*>(Xbf)[i] = reinterpret_cast<const uint4*>(xnb)[i];
    for (int i = tid; i < (DIMSZ * DIMSZ) / 8; i += 256)
      reinterpret_cast<uint4*>(WbT)[i] = reinterpret_cast<const uint4*>(wpre)[i];
#endif
  } else {
    const float* xn = x + (size_t)n * MSEQ * DIMSZ;
    for (int i = tid; i < MSEQ * DIMSZ / 4; i += 256) {
      const float4 v = reinterpret_cast<const float4*>(xn)[i];
      union { bf16 h[4]; uint2 u; } pk;
      pk.h[0] = (bf16)v.x; pk.h[1] = (bf16)v.y; pk.h[2] = (bf16)v.z; pk.h[3] = (bf16)v.w;
      reinterpret_cast<uint2*>(Xbf)[i] = pk.u;
    }
    for (int i = tid; i < DIMSZ * DIMSZ; i += 256) {
      const int e = i >> 7, f = i & 127;
      float s = 0.f;
#pragma unroll
      for (int h = 0; h < 4; ++h) s += rw[((h * DIMSZ) + e) * DIMSZ + f];
      WbT[f * DIMSZ + e] = (bf16)(s * K_W_SCALE);
    }
  }
  __syncthreads();

  // Per-lane fragment geometry (ISA 7.12.2, 16-bit operands, wave32)
  const int rh    = wave >> 2;          // row half   (2 x 16-row blocks)
  const int cq    = wave & 3;           // col quarter
  const int lm    = lane & 15;
  const int hi16  = lane >> 4;
  const int kb    = hi16 * 8;           // A-fragment K base
  const int khalf = hi16 * 16;          // unused; B fragments also use kb

  // ---- GEMM1: Y(64x128) = Xtile(64x128) @ Wbar ; wave -> 32 x 32 tile (2x2 WMMA)
  {
    v8f yacc[2][2] = {};
#pragma unroll
    for (int e0 = 0; e0 < DIMSZ; e0 += 32) {
      v16bf a[2];
#pragma unroll
      for (int r = 0; r < 2; ++r) {
        const bf16* Ar = Xbf + (size_t)(mt * TM + rh * 32 + r * 16 + lm) * DIMSZ;
        a[r] = ld_frag(Ar + e0 + kb, Ar + e0 + kb + 16);
      }
#pragma unroll
      for (int t = 0; t < 2; ++t) {
        const bf16* Brow = WbT + (size_t)(cq * 32 + t * 16 + lm) * DIMSZ;
        v16bf b = ld_frag(Brow + e0 + kb, Brow + e0 + kb + 16);
#pragma unroll
        for (int r = 0; r < 2; ++r) yacc[r][t] = wmma_bf16(a[r], b, yacc[r][t]);
      }
    }
#pragma unroll
    for (int r = 0; r < 2; ++r)
#pragma unroll
      for (int t = 0; t < 2; ++t)
#pragma unroll
        for (int v = 0; v < 8; ++v)
          Ybf[(rh * 32 + r * 16 + v + hi16 * 8) * DIMSZ + (cq * 32 + t * 16 + lm)] =
              (bf16)yacc[r][t][v];
  }
  __syncthreads();   // Ybf ready; WbT region may now be overwritten by S

  // ---- GEMM2: S(64x512) = Y(64x128) @ X^T ; wave -> 32 x 128 tile (2x8 WMMA)
  {
    v8f sacc[2][8] = {};
#pragma unroll
    for (int e0 = 0; e0 < DIMSZ; e0 += 32) {
      v16bf a[2];
#pragma unroll
      for (int r = 0; r < 2; ++r) {
        const bf16* Ar = Ybf + (size_t)(rh * 32 + r * 16 + lm) * DIMSZ;
        a[r] = ld_frag(Ar + e0 + kb, Ar + e0 + kb + 16);
      }
#pragma unroll
      for (int t = 0; t < 8; ++t) {
        const bf16* Brow = Xbf + (size_t)(cq * 128 + t * 16 + lm) * DIMSZ;
        v16bf b = ld_frag(Brow + e0 + kb, Brow + e0 + kb + 16);
#pragma unroll
        for (int r = 0; r < 2; ++r) sacc[r][t] = wmma_bf16(a[r], b, sacc[r][t]);
      }
    }
#pragma unroll
    for (int r = 0; r < 2; ++r)
#pragma unroll
      for (int t = 0; t < 8; ++t)
#pragma unroll
        for (int v = 0; v < 8; ++v)
          S[(rh * 32 + r * 16 + v + hi16 * 8) * MSEQ + (cq * 128 + t * 16 + lm)] =
              sacc[r][t][v];
  }
  __syncthreads();

  // ---- Softmax row stats: 4 threads per row
  {
    const int row = tid >> 2, part = tid & 3;
    const float* Sr = S + row * MSEQ + part * 128;
    float mx = -3.402823466e38f;
    for (int i = 0; i < 128; ++i) mx = fmaxf(mx, Sr[i]);
    mx = fmaxf(mx, __shfl_xor(mx, 1));
    mx = fmaxf(mx, __shfl_xor(mx, 2));
    float sm = 0.f;
    for (int i = 0; i < 128; ++i) sm += __expf(Sr[i] - mx);
    sm += __shfl_xor(sm, 1);
    sm += __shfl_xor(sm, 2);
    if (part == 0) { rowmax[row] = mx; rowsum[row] = 1.0f / sm; }
  }
  __syncthreads();

  // ---- Column sums of normalized weights -> G[n][k] (atomic accumulate over m-tiles)
  for (int k = tid; k < MSEQ; k += 256) {
    float acc = 0.f;
    for (int m = 0; m < TM; ++m)
      acc += __expf(S[m * MSEQ + k] - rowmax[m]) * rowsum[m];
    atomicAdd(&G[n * MSEQ + k], acc);
  }
}

__global__ __launch_bounds__(128)
void routing_final_kernel(const float* __restrict__ x,
                          const float* __restrict__ G,
                          const float* __restrict__ pw,
                          const float* __restrict__ pb,
                          float* __restrict__ out) {
  __shared__ float o[DIMSZ];
  const int n = blockIdx.x, d = threadIdx.x;
  const float* xn = x + (size_t)n * MSEQ * DIMSZ;
  const float* g  = G + n * MSEQ;
  float acc = 0.f;
  for (int k = 0; k < MSEQ; ++k) acc += g[k] * xn[k * DIMSZ + d];
  o[d] = acc * (1.0f / (float)MSEQ);
  __syncthreads();
  float r = pb[d];
  const float* pwr = pw + d * DIMSZ;   // out[i] = sum_e o[e] * proj_w[i][e] + b[i]
  for (int e = 0; e < DIMSZ; ++e) r += o[e] * pwr[e];
  out[n * DIMSZ + d] = r;
}

extern "C" void kernel_launch(void* const* d_in, const int* in_sizes, int n_in,
                              void* d_out, int out_size, void* d_ws, size_t ws_size,
                              hipStream_t stream) {
  const float* x  = (const float*)d_in[0];   // [128, 512, 128]
  const float* rw = (const float*)d_in[1];   // [4, 128, 128]
  const float* pw = (const float*)d_in[2];   // [128, 128]
  const float* pb = (const float*)d_in[3];   // [128]
  float* out = (float*)d_out;                // [128, 128]

  char* ws = (char*)d_ws;
  float* G   = (float*)(ws + WS_G_OFF);      // [128, 512]
  bf16* wbt  = (bf16*)(ws + WS_WBT_OFF);     // [128, 128]
  bf16* xb   = (bf16*)(ws + WS_XB_OFF);      // [128*512*128]
  const int use_pre = (ws_size >= WS_NEED) ? 1 : 0;

  (void)hipMemsetAsync(G, 0, (size_t)NB * MSEQ * sizeof(float), stream);
  if (use_pre) {
    convert_x_kernel<<<dim3((NB * MSEQ * DIMSZ / 4) / 256), dim3(256), 0, stream>>>(x, xb);
    build_wbt_kernel<<<dim3((DIMSZ * DIMSZ) / 256), dim3(256), 0, stream>>>(rw, wbt);
  }
  routing_scores_kernel<<<dim3(NB * TILES), dim3(256), SMEM_BYTES, stream>>>(
      x, rw, xb, wbt, G, use_pre);
  routing_final_kernel<<<dim3(NB), dim3(128), 0, stream>>>(x, G, pw, pb, out);
}
